// UNet_SCN_classifier_22419729285189
// MI455X (gfx1250) — matrix operand order repaired and checked by reading, not compiled, not run on hardware
//
#include <hip/hip_runtime.h>
#include <hip/hip_bf16.h>
#include <cstdint>

// ---------------------------------------------------------------------------
// Dense UNet-SCN on MI455X (gfx1250, wave32).
//
// Every conv (normal, strided, transpose) is an implicit GEMM on
// v_wmma_f32_16x16x32_bf16. Activations are channels-last bf16 (C padded to
// 32) so im2col A-rows are contiguous 64B chunks; weights are pre-swizzled
// into the exact B-fragment layout. The K-loop is double-buffered in both
// registers and LDS: next step's global loads issue before this step's
// WMMAs, so the matrix pipe never waits on LOADcnt. BN (training stats) is a
// deterministic two-pass reduction; basic-block conv biases cancel under
// training-mode BN and are dropped. f32 accumulation throughout.
// ---------------------------------------------------------------------------

typedef __attribute__((ext_vector_type(16))) __bf16 v16bf;
typedef __attribute__((ext_vector_type(8)))  float  v8f;

union FragBF { uint4 q[2]; v16bf v; };

__device__ __forceinline__ float bf2f(__bf16 b) {
  unsigned short h = __builtin_bit_cast(unsigned short, b);
  unsigned u = ((unsigned)h) << 16;
  return __builtin_bit_cast(float, u);
}
__device__ __forceinline__ __bf16 f2bf(float f) {
  unsigned u = __builtin_bit_cast(unsigned, f);
  unsigned r = u + 0x7FFFu + ((u >> 16) & 1u);   // RNE
  unsigned short h = (unsigned short)(r >> 16);
  return __builtin_bit_cast(__bf16, h);
}

// ---------------------------------------------------------------------------
// Implicit-GEMM conv. Block = 256 thr (8 waves) -> 64(M) x 128(N) out tile.
// Each wave: 2x2 tiles of 16x16, K-steps of 32 over fused (tap, cin-block).
// transpose==1 implements the linear transpose of the stride-2 conv
// (SparseInverseConv3d rulebook reversal on a dense grid).
// ---------------------------------------------------------------------------
#define APITCH 40   // halfs per A row in LDS (80B, 16B aligned, bank-skewed)

__global__ __launch_bounds__(256) void conv_wmma(
    const __bf16* __restrict__ X,   // [inP][Cs] channels-last bf16
    const __bf16* __restrict__ Wp,  // [27][nKb][Cout][32] bf16 fragments
    float* __restrict__ Y,          // [outP][Cout] f32
    int Cs, int l2nKb, int Cout,
    int inD, int inH, int inW,
    int lw, int lh,                 // log2 of output W, H
    int stride, int pd, int ph, int pw,
    int transpose)
{
  // double-buffered tiles: one barrier per K-step
  __shared__ __align__(16) __bf16 Alds[2][64 * APITCH];
  __shared__ __align__(16) __bf16 Blds[2][128 * 32];

  const int t     = threadIdx.x;
  const int lane  = t & 31;
  const int wave  = t >> 5;
  const int waveM = wave & 1;      // 2 M groups of 32
  const int waveN = wave >> 1;     // 4 N groups of 32
  const int lhalf = lane >> 4;
  const int lrow  = lane & 15;
  const int mBase = blockIdx.x * 64;
  const int nBase = blockIdx.y * 128;

  // A-gather role: one uint4 (8 halfs) per thread per K-step
  const int arow   = t >> 2;       // 0..63 output row within tile
  const int achunk = t & 3;        // 0..3  (8-half chunk within K=32)
  const int m  = mBase + arow;
  const int od = m >> (lw + lh);
  const int oh = (m >> lw) & ((1 << lh) - 1);
  const int ow = m & ((1 << lw) - 1);
  const int kbMask = (1 << l2nKb) - 1;

  // fetch (tap,kb) step s into registers
  auto fetch = [&](int s, uint4& av, uint4& b0, uint4& b1) {
    const int tap = s >> l2nKb;
    const int kb  = s & kbMask;
    const int kd = tap / 9;
    const int kr = tap - kd * 9;
    const int kh = kr / 3;
    const int kw = kr - kh * 3;
    bool valid;
    int  src;
    if (!transpose) {
      int id = od * stride + kd - pd;
      int ih = oh * stride + kh - ph;
      int iw = ow * stride + kw - pw;
      valid = ((unsigned)id < (unsigned)inD) & ((unsigned)ih < (unsigned)inH) &
              ((unsigned)iw < (unsigned)inW);
      src = (id * inH + ih) * inW + iw;
    } else {
      int td = od + pd - kd, th = oh + ph - kh, tw = ow + pw - kw;
      bool ok = (td >= 0) & (th >= 0) & (tw >= 0) & !(((td | th | tw) & 1));
      int id = td >> 1, ih = th >> 1, iw = tw >> 1;
      valid = ok & (id < inD) & (ih < inH) & (iw < inW);
      src = (id * inH + ih) * inW + iw;
    }
    av.x = av.y = av.z = av.w = 0u;
    if (valid)
      av = *reinterpret_cast<const uint4*>(X + (size_t)src * Cs + kb * 32 + achunk * 8);
    const uint4* wg = reinterpret_cast<const uint4*>(
        Wp + (((size_t)((tap << l2nKb) + kb) * Cout + nBase) * 32));
    b0 = wg[t];
    b1 = wg[t + 256];
  };

  v8f acc[2][2];
  #pragma unroll
  for (int i = 0; i < 2; ++i)
    #pragma unroll
    for (int j = 0; j < 2; ++j) acc[i][j] = (v8f){0.f,0.f,0.f,0.f,0.f,0.f,0.f,0.f};

  const int S = 27 << l2nKb;

  // prologue: fill buffer 0
  uint4 av, b0, b1;
  fetch(0, av, b0, b1);
  *reinterpret_cast<uint4*>(&Alds[0][arow * APITCH + achunk * 8]) = av;
  reinterpret_cast<uint4*>(Blds[0])[t]       = b0;
  reinterpret_cast<uint4*>(Blds[0])[t + 256] = b1;

  for (int s = 0; s < S; ++s) {
    const int cur = s & 1, nxt = cur ^ 1;
    __syncthreads();    // buf[cur] complete for all waves

    // fragments for this step (ISA wave32 VGPR layouts)
    FragBF a[2], b[2];
    #pragma unroll
    for (int mt = 0; mt < 2; ++mt) {
      const __bf16* ap = &Alds[cur][(waveM * 32 + mt * 16 + lrow) * APITCH + lhalf * 8];
      a[mt].q[0] = *reinterpret_cast<const uint4*>(ap);        // K = lhalf*8 .. +8
      a[mt].q[1] = *reinterpret_cast<const uint4*>(ap + 16);   // K = 16+lhalf*8 ..
    }
    #pragma unroll
    for (int nt = 0; nt < 2; ++nt) {
      const __bf16* bp = &Blds[cur][(waveN * 32 + nt * 16 + lrow) * 32 + lhalf * 8];
      b[nt].q[0] = *reinterpret_cast<const uint4*>(bp);        // K = lhalf*8 .. +8
      b[nt].q[1] = *reinterpret_cast<const uint4*>(bp + 16);   // K = 16+lhalf*8 ..
    }

    // issue next step's global loads before the matrix ops (latency hiding)
    if (s + 1 < S) fetch(s + 1, av, b0, b1);

    #pragma unroll
    for (int mt = 0; mt < 2; ++mt)
      #pragma unroll
      for (int nt = 0; nt < 2; ++nt)
        acc[mt][nt] = __builtin_amdgcn_wmma_f32_16x16x32_bf16(
            false, a[mt].v, false, b[nt].v, (short)0, acc[mt][nt], false, false);

    // drain prefetched registers into the alternate buffer (wait hidden by WMMAs)
    if (s + 1 < S) {
      *reinterpret_cast<uint4*>(&Alds[nxt][arow * APITCH + achunk * 8]) = av;
      reinterpret_cast<uint4*>(Blds[nxt])[t]       = b0;
      reinterpret_cast<uint4*>(Blds[nxt])[t + 256] = b1;
    }
  }

  // ---- store D tiles, channels-last (coalesced across lrow) ----
  #pragma unroll
  for (int mt = 0; mt < 2; ++mt)
    #pragma unroll
    for (int nt = 0; nt < 2; ++nt)
      #pragma unroll
      for (int r = 0; r < 8; ++r) {
        int mm = mBase + waveM * 32 + mt * 16 + r + lhalf * 8;
        int nn = nBase + waveN * 32 + nt * 16 + lrow;
        Y[(size_t)mm * Cout + nn] = acc[mt][nt][r];
      }
}

// ---------------------------------------------------------------------------
// BN training stats: deterministic two-pass tree (no FP atomics).
// ---------------------------------------------------------------------------
__global__ void bn_partial(const float* __restrict__ Y, float* __restrict__ part,
                           int C, int P, int chunk) {
  int c = blockIdx.y * 256 + threadIdx.x;
  if (c >= C) return;
  int p0 = blockIdx.x * chunk;
  int p1 = p0 + chunk; if (p1 > P) p1 = P;
  float s = 0.f, s2 = 0.f;
  for (int p = p0; p < p1; ++p) {
    float v = Y[(size_t)p * C + c];
    s += v; s2 += v * v;
  }
  part[((size_t)blockIdx.x * C + c) * 2]     = s;
  part[((size_t)blockIdx.x * C + c) * 2 + 1] = s2;
}

__global__ void bn_finalize(const float* __restrict__ part, int G, int C, int P,
                            const float* __restrict__ g, const float* __restrict__ b,
                            float* __restrict__ scale, float* __restrict__ shift) {
  int c = blockIdx.x * 256 + threadIdx.x;
  if (c >= C) return;
  float s = 0.f, s2 = 0.f;
  for (int i = 0; i < G; ++i) {
    s  += part[((size_t)i * C + c) * 2];
    s2 += part[((size_t)i * C + c) * 2 + 1];
  }
  float inv = 1.f / (float)P;
  float mu  = s * inv;
  float var = s2 * inv - mu * mu;          // biased var (jnp.var default)
  float sc  = g[c] * rsqrtf(var + 1e-4f);
  scale[c] = sc;
  shift[c] = b[c] - mu * sc;
}

// mode 0: out = bf16(tanh(y*sc+sh))
// mode 1: out = bf16(tanh(y*sc+sh + res))                (basic-block tail)
// mode 2: out = bf16(tanh(y*sc+sh) + cat[2c] + cat[2c+1]) (merge + chan-red)
__global__ void bn_apply(const float* __restrict__ Y,
                         const float* __restrict__ scale, const float* __restrict__ shift,
                         const __bf16* __restrict__ res, const __bf16* __restrict__ cat,
                         __bf16* __restrict__ out, int l2C, size_t total, int mode) {
  size_t i = (size_t)blockIdx.x * 256 + threadIdx.x;
  if (i >= total) return;
  int    c = (int)(i & ((1u << l2C) - 1));
  size_t p = i >> l2C;
  float v = Y[i] * scale[c] + shift[c];
  float o;
  if (mode == 0) {
    o = tanhf(v);
  } else if (mode == 1) {
    o = tanhf(v + bf2f(res[i]));
  } else {
    size_t base = (p << (l2C + 1)) + ((size_t)c << 1);
    o = tanhf(v) + bf2f(cat[base]) + bf2f(cat[base + 1]);
  }
  out[i] = f2bf(o);
}

__global__ void concat_cl(const __bf16* __restrict__ a, const __bf16* __restrict__ b,
                          __bf16* __restrict__ out, int l2C, size_t total) {
  size_t i = (size_t)blockIdx.x * 256 + threadIdx.x;
  if (i >= total) return;
  int C = 1 << l2C;
  int c = (int)(i & ((2u << l2C) - 1));
  size_t p = i >> (l2C + 1);
  out[i] = (c < C) ? a[(p << l2C) + c] : b[(p << l2C) + (c - C)];
}

// f32 NCDHW (C=4) -> channels-last bf16 padded to 32 channels
__global__ void cvt_input(const float* __restrict__ x, __bf16* __restrict__ out, int P) {
  int i = blockIdx.x * 256 + threadIdx.x;
  if (i >= P * 32) return;
  int c = i & 31, p = i >> 5;
  out[i] = (c < 4) ? f2bf(x[(size_t)c * P + p]) : f2bf(0.f);
}

// Pre-swizzle conv weights into B-fragment layout [tap][kblk][O][32] bf16.
// trans==0: w is (O, Kreal, 3,3,3).  trans==1: w is (Kreal, O, 3,3,3).
__global__ void prep_w(const float* __restrict__ w, __bf16* __restrict__ out,
                       int l2O, int l2nKb, int Kreal, int trans, size_t total) {
  size_t i = (size_t)blockIdx.x * 256 + threadIdx.x;
  if (i >= total) return;
  int ki  = (int)(i & 31);
  int o   = (int)((i >> 5) & ((1u << l2O) - 1));
  int kb  = (int)((i >> (5 + l2O)) & ((1u << l2nKb) - 1));
  int tap = (int)(i >> (5 + l2O + l2nKb));
  int kg  = kb * 32 + ki;
  float v = 0.f;
  if (kg < Kreal)
    v = trans ? w[((size_t)kg * (1 << l2O) + o) * 27 + tap]
              : w[((size_t)o * Kreal + kg) * 27 + tap];
  out[i] = f2bf(v);
}

// logits[o][p] = sum_c up1[p][c]*W[o][c] + b[o]   (128 -> 10)
__global__ __launch_bounds__(256) void classifier_k(
    const __bf16* __restrict__ up, const float* __restrict__ W,
    const float* __restrict__ bias, float* __restrict__ out, int P) {
  __shared__ float Wl[10 * 128];
  __shared__ float bl[10];
  for (int i = threadIdx.x; i < 1280; i += 256) Wl[i] = W[i];
  if (threadIdx.x < 10) bl[threadIdx.x] = bias[threadIdx.x];
  __syncthreads();
  int p = blockIdx.x * 256 + threadIdx.x;
  if (p >= P) return;
  uint4 rq[16];
  const uint4* rp = reinterpret_cast<const uint4*>(up + (size_t)p * 128);
  #pragma unroll
  for (int j = 0; j < 16; ++j) rq[j] = rp[j];
  const __bf16* rh = reinterpret_cast<const __bf16*>(rq);
  float acc[10];
  #pragma unroll
  for (int o = 0; o < 10; ++o) acc[o] = 0.f;
  for (int c = 0; c < 128; ++c) {
    float xv = bf2f(rh[c]);
    #pragma unroll
    for (int o = 0; o < 10; ++o) acc[o] += xv * Wl[o * 128 + c];
  }
  #pragma unroll
  for (int o = 0; o < 10; ++o) out[(size_t)o * P + p] = acc[o] + bl[o];
}

// ---------------------------------------------------------------------------
// Host orchestration
// ---------------------------------------------------------------------------
namespace {
struct DimL { int D, H, W, P, lw, lh; };
static inline int ilog2i(int v) { int l = 0; while ((1 << l) < v) ++l; return l; }
}

extern "C" void kernel_launch(void* const* d_in, const int* in_sizes, int n_in,
                              void* d_out, int out_size, void* d_ws, size_t ws_size,
                              hipStream_t stream) {
  (void)in_sizes; (void)n_in; (void)out_size; (void)ws_size;

  // ---- input mapping: jax.tree_util leaf order (dict keys sorted) ----
  // top: 'params' then 'x'; params keys: cls, conv1, conv2, conv3, conv4,
  // conv5, conv_input, inv2, inv3, inv4, m1..m4, t1..t4.
  // lyr/ilyr dict -> (b, g, w); bb conv dict -> (b, bias, g, w); cls -> (b, w).
  int idx = 0;
  auto F = [&](int i) { return (const float*)d_in[i]; };
  struct Lyr { const float *b, *g, *w; };
  auto getL = [&]() { Lyr L; L.b = F(idx++); L.g = F(idx++); L.w = F(idx++); return L; };
  struct BB { Lyr c1, c2; };
  auto getBB = [&]() {            // (b, bias, g, w) x2 ; bias is a BN no-op
    BB r;
    r.c1.b = F(idx++); idx++; r.c1.g = F(idx++); r.c1.w = F(idx++);
    r.c2.b = F(idx++); idx++; r.c2.g = F(idx++); r.c2.w = F(idx++);
    return r;
  };

  const float* cls_b = F(idx++);
  const float* cls_w = F(idx++);
  Lyr conv1 = getL();
  Lyr conv2_[3]; for (int i = 0; i < 3; ++i) conv2_[i] = getL();
  Lyr conv3_[3]; for (int i = 0; i < 3; ++i) conv3_[i] = getL();
  Lyr conv4_[3]; for (int i = 0; i < 3; ++i) conv4_[i] = getL();
  Lyr conv5  = getL();
  Lyr convin = getL();
  Lyr inv2 = getL(), inv3 = getL(), inv4 = getL();
  Lyr m1 = getL(), m2 = getL(), m3 = getL(), m4 = getL();
  BB t1 = getBB(), t2 = getBB(), t3 = getBB(), t4 = getBB();
  const float* x_in = F(idx++);

  // ---- workspace bump allocator ----
  size_t off = 0;
  auto alloc = [&](size_t bytes) {
    bytes = (bytes + 255) & ~(size_t)255;
    void* p = (char*)d_ws + off;
    off += bytes;
    return p;
  };

  const DimL L1{32, 64, 64, 131072, 6, 6};
  const DimL L2{16, 32, 32, 16384, 5, 5};
  const DimL L3{8, 16, 16, 2048, 4, 4};
  const DimL L4{3, 8, 8, 192, 3, 3};
  const int P1 = L1.P, P2 = L2.P, P3 = L3.P, P4 = L4.P;

  float* Ybuf  = (float*)alloc((size_t)P1 * 128 * 4);      // f32 conv output (max)
  float* partb = (float*)alloc((size_t)128 * 1024 * 2 * 4);
  float* scb   = (float*)alloc(1024 * 4);
  float* shb   = (float*)alloc(1024 * 4);

  __bf16* X0  = (__bf16*)alloc((size_t)P1 * 32 * 2);
  __bf16* x1  = (__bf16*)alloc((size_t)P1 * 128 * 2);
  __bf16* s1a = (__bf16*)alloc((size_t)P1 * 128 * 2);
  __bf16* s1b = (__bf16*)alloc((size_t)P1 * 128 * 2);
  __bf16* u1  = (__bf16*)alloc((size_t)P1 * 128 * 2);
  __bf16* x2  = (__bf16*)alloc((size_t)P2 * 256 * 2);
  __bf16* s2a = (__bf16*)alloc((size_t)P2 * 256 * 2);
  __bf16* s2b = (__bf16*)alloc((size_t)P2 * 256 * 2);
  __bf16* u2  = (__bf16*)alloc((size_t)P2 * 256 * 2);
  __bf16* x3  = (__bf16*)alloc((size_t)P3 * 512 * 2);
  __bf16* s3a = (__bf16*)alloc((size_t)P3 * 512 * 2);
  __bf16* s3b = (__bf16*)alloc((size_t)P3 * 512 * 2);
  __bf16* u3  = (__bf16*)alloc((size_t)P3 * 512 * 2);
  __bf16* x4  = (__bf16*)alloc((size_t)P4 * 512 * 2);
  __bf16* s4a = (__bf16*)alloc((size_t)P4 * 512 * 2);
  __bf16* s4b = (__bf16*)alloc((size_t)P4 * 512 * 2);
  __bf16* cat1 = (__bf16*)alloc((size_t)P1 * 256 * 2);
  __bf16* cat2 = (__bf16*)alloc((size_t)P2 * 512 * 2);
  __bf16* cat3 = (__bf16*)alloc((size_t)P3 * 1024 * 2);
  __bf16* cat4 = (__bf16*)alloc((size_t)P4 * 1024 * 2);

  // ---- weight pre-swizzle ----
  auto prepW = [&](const float* w, int O, int Kreal, bool trans) -> __bf16* {
    int Kpad = (Kreal + 31) & ~31;
    int nKb = Kpad / 32;
    size_t total = (size_t)27 * nKb * O * 32;
    __bf16* out = (__bf16*)alloc(total * 2);
    prep_w<<<(unsigned)((total + 255) / 256), 256, 0, stream>>>(
        w, out, ilog2i(O), ilog2i(nKb), Kreal, trans ? 1 : 0, total);
    return out;
  };

  __bf16* Wci   = prepW(convin.w, 128, 4, false);
  __bf16* Wc1   = prepW(conv1.w, 128, 128, false);
  __bf16* Wc2[3] = {prepW(conv2_[0].w, 256, 128, false),
                    prepW(conv2_[1].w, 256, 256, false),
                    prepW(conv2_[2].w, 256, 256, false)};
  __bf16* Wc3[3] = {prepW(conv3_[0].w, 512, 256, false),
                    prepW(conv3_[1].w, 512, 512, false),
                    prepW(conv3_[2].w, 512, 512, false)};
  __bf16* Wc4[3] = {prepW(conv4_[0].w, 512, 512, false),
                    prepW(conv4_[1].w, 512, 512, false),
                    prepW(conv4_[2].w, 512, 512, false)};
  __bf16 *Wt4a = prepW(t4.c1.w, 512, 512, false), *Wt4b = prepW(t4.c2.w, 512, 512, false);
  __bf16 *Wm4  = prepW(m4.w, 512, 1024, false);
  __bf16 *Wi4  = prepW(inv4.w, 512, 512, true);   // N=c_bot, K=c_top
  __bf16 *Wt3a = prepW(t3.c1.w, 512, 512, false), *Wt3b = prepW(t3.c2.w, 512, 512, false);
  __bf16 *Wm3  = prepW(m3.w, 512, 1024, false);
  __bf16 *Wi3  = prepW(inv3.w, 256, 512, true);
  __bf16 *Wt2a = prepW(t2.c1.w, 256, 256, false), *Wt2b = prepW(t2.c2.w, 256, 256, false);
  __bf16 *Wm2  = prepW(m2.w, 256, 512, false);
  __bf16 *Wi2  = prepW(inv2.w, 128, 256, true);
  __bf16 *Wt1a = prepW(t1.c1.w, 128, 128, false), *Wt1b = prepW(t1.c2.w, 128, 128, false);
  __bf16 *Wm1  = prepW(m1.w, 128, 256, false);
  __bf16 *Wc5  = prepW(conv5.w, 128, 128, false);

  // ---- conv + BN + epilogue helper ----
  auto convBN = [&](const __bf16* Xa, int Cs, const __bf16* Wp, int nKb, int Cout,
                    DimL in, DimL out, int stride, int pd, int ph, int pw, int trans,
                    const float* g, const float* bprm, int mode,
                    const __bf16* res, const __bf16* cat, __bf16* outAct) {
    dim3 grid(out.P / 64, Cout / 128);
    conv_wmma<<<grid, 256, 0, stream>>>(Xa, Wp, Ybuf, Cs, ilog2i(nKb), Cout,
                                        in.D, in.H, in.W, out.lw, out.lh,
                                        stride, pd, ph, pw, trans);
    int G = (out.P >= 1024) ? ((out.P / 1024 < 128) ? out.P / 1024 : 128) : 1;
    int chunk = (out.P + G - 1) / G;
    dim3 g1(G, (Cout + 255) / 256);
    bn_partial<<<g1, 256, 0, stream>>>(Ybuf, partb, Cout, out.P, chunk);
    bn_finalize<<<(Cout + 255) / 256, 256, 0, stream>>>(partb, G, Cout, out.P, g, bprm, scb, shb);
    size_t total = (size_t)out.P * Cout;
    bn_apply<<<(unsigned)((total + 255) / 256), 256, 0, stream>>>(
        Ybuf, scb, shb, res, cat, outAct, ilog2i(Cout), total, mode);
  };
  auto docat = [&](const __bf16* a, const __bf16* b, __bf16* o, int C, int P) {
    size_t total = (size_t)P * 2 * C;
    concat_cl<<<(unsigned)((total + 255) / 256), 256, 0, stream>>>(a, b, o, ilog2i(C), total);
  };

  // ---- forward ----
  cvt_input<<<(P1 * 32) / 256, 256, 0, stream>>>(x_in, X0, P1);

  convBN(X0, 32, Wci, 1, 128, L1, L1, 1, 1, 1, 1, 0, convin.g, convin.b, 0, nullptr, nullptr, s1a);
  convBN(s1a, 128, Wc1, 4, 128, L1, L1, 1, 1, 1, 1, 0, conv1.g, conv1.b, 0, nullptr, nullptr, x1);

  convBN(x1, 128, Wc2[0], 4, 256, L1, L2, 2, 1, 1, 1, 0, conv2_[0].g, conv2_[0].b, 0, nullptr, nullptr, s2a);
  convBN(s2a, 256, Wc2[1], 8, 256, L2, L2, 1, 1, 1, 1, 0, conv2_[1].g, conv2_[1].b, 0, nullptr, nullptr, s2b);
  convBN(s2b, 256, Wc2[2], 8, 256, L2, L2, 1, 1, 1, 1, 0, conv2_[2].g, conv2_[2].b, 0, nullptr, nullptr, x2);

  convBN(x2, 256, Wc3[0], 8, 512, L2, L3, 2, 1, 1, 1, 0, conv3_[0].g, conv3_[0].b, 0, nullptr, nullptr, s3a);
  convBN(s3a, 512, Wc3[1], 16, 512, L3, L3, 1, 1, 1, 1, 0, conv3_[1].g, conv3_[1].b, 0, nullptr, nullptr, s3b);
  convBN(s3b, 512, Wc3[2], 16, 512, L3, L3, 1, 1, 1, 1, 0, conv3_[2].g, conv3_[2].b, 0, nullptr, nullptr, x3);

  convBN(x3, 512, Wc4[0], 16, 512, L3, L4, 2, 0, 1, 1, 0, conv4_[0].g, conv4_[0].b, 0, nullptr, nullptr, s4a);
  convBN(s4a, 512, Wc4[1], 16, 512, L4, L4, 1, 1, 1, 1, 0, conv4_[1].g, conv4_[1].b, 0, nullptr, nullptr, s4b);
  convBN(s4b, 512, Wc4[2], 16, 512, L4, L4, 1, 1, 1, 1, 0, conv4_[2].g, conv4_[2].b, 0, nullptr, nullptr, x4);

  // UR4 (lateral = bottom = x4)
  convBN(x4, 512, Wt4a, 16, 512, L4, L4, 1, 1, 1, 1, 0, t4.c1.g, t4.c1.b, 0, nullptr, nullptr, s4a);
  convBN(s4a, 512, Wt4b, 16, 512, L4, L4, 1, 1, 1, 1, 0, t4.c2.g, t4.c2.b, 1, x4, nullptr, s4b);
  docat(x4, s4b, cat4, 512, P4);
  convBN(cat4, 1024, Wm4, 32, 512, L4, L4, 1, 1, 1, 1, 0, m4.g, m4.b, 2, nullptr, cat4, s4a);
  convBN(s4a, 512, Wi4, 16, 512, L4, L3, 2, 0, 1, 1, 1, inv4.g, inv4.b, 0, nullptr, nullptr, u3);

  // UR3
  convBN(x3, 512, Wt3a, 16, 512, L3, L3, 1, 1, 1, 1, 0, t3.c1.g, t3.c1.b, 0, nullptr, nullptr, s3a);
  convBN(s3a, 512, Wt3b, 16, 512, L3, L3, 1, 1, 1, 1, 0, t3.c2.g, t3.c2.b, 1, x3, nullptr, s3b);
  docat(u3, s3b, cat3, 512, P3);
  convBN(cat3, 1024, Wm3, 32, 512, L3, L3, 1, 1, 1, 1, 0, m3.g, m3.b, 2, nullptr, cat3, s3a);
  convBN(s3a, 512, Wi3, 16, 256, L3, L2, 2, 1, 1, 1, 1, inv3.g, inv3.b, 0, nullptr, nullptr, u2);

  // UR2
  convBN(x2, 256, Wt2a, 8, 256, L2, L2, 1, 1, 1, 1, 0, t2.c1.g, t2.c1.b, 0, nullptr, nullptr, s2a);
  convBN(s2a, 256, Wt2b, 8, 256, L2, L2, 1, 1, 1, 1, 0, t2.c2.g, t2.c2.b, 1, x2, nullptr, s2b);
  docat(u2, s2b, cat2, 256, P2);
  convBN(cat2, 512, Wm2, 16, 256, L2, L2, 1, 1, 1, 1, 0, m2.g, m2.b, 2, nullptr, cat2, s2a);
  convBN(s2a, 256, Wi2, 8, 128, L2, L1, 2, 1, 1, 1, 1, inv2.g, inv2.b, 0, nullptr, nullptr, u1);

  // UR1 (final level: subm conv5 instead of inverse conv)
  convBN(x1, 128, Wt1a, 4, 128, L1, L1, 1, 1, 1, 1, 0, t1.c1.g, t1.c1.b, 0, nullptr, nullptr, s1b);
  convBN(s1b, 128, Wt1b, 4, 128, L1, L1, 1, 1, 1, 1, 0, t1.c2.g, t1.c2.b, 1, x1, nullptr, s1a);
  docat(u1, s1a, cat1, 128, P1);
  convBN(cat1, 256, Wm1, 8, 128, L1, L1, 1, 1, 1, 1, 0, m1.g, m1.b, 2, nullptr, cat1, s1b);
  convBN(s1b, 128, Wc5, 4, 128, L1, L1, 1, 1, 1, 1, 0, conv5.g, conv5.b, 0, nullptr, nullptr, u1);

  // classifier 128 -> 10, output NCDHW f32
  classifier_k<<<P1 / 256, 256, 0, stream>>>(u1, cls_w, cls_b, (float*)d_out, P1);
}
